// StandardRoIHeadWarper_28260884807773
// MI455X (gfx1250) — compile-verified, matrix operand order checked by reading, not run on hardware
//
#include <hip/hip_runtime.h>
#include <cstdint>

typedef __attribute__((ext_vector_type(16))) __bf16 v16bf;
typedef __attribute__((ext_vector_type(8)))  float  v8f;

#define NPROP 1000
#define MPAD  1024
#define D_IN  12544   /* 256 * 7 * 7 */
#define FC    1024
#define NCLS  81
#define CLSP  128
#define REGP  384
#define PGRID 14

__device__ __forceinline__ unsigned short f2bf(float f) {
  unsigned u = __float_as_uint(f);
  unsigned r = u + 0x7fffu + ((u >> 16) & 1u);   // round to nearest even
  return (unsigned short)(r >> 16);
}

// Direct global -> LDS async copy (16B per lane), tracked by ASYNCcnt.
__device__ __forceinline__ void async_copy16(unsigned ldsoff, const unsigned short* g) {
  unsigned long long ga = (unsigned long long)(uintptr_t)g;
  asm volatile("global_load_async_to_lds_b128 %0, %1, off"
               :: "v"(ldsoff), "v"(ga) : "memory");
}
__device__ __forceinline__ void wait_async_le3() {
  asm volatile("s_wait_asynccnt 0x3" ::: "memory");
}
__device__ __forceinline__ void wait_async_0() {
  asm volatile("s_wait_asynccnt 0x0" ::: "memory");
}

// ---------------------------------------------------------------------------
// Weight conversion f32 -> bf16 with zero row padding
// ---------------------------------------------------------------------------
__global__ void cvt_pad_bf16(const float* __restrict__ src, unsigned short* __restrict__ dst,
                             int rows, int cols, long long total) {
  long long i = (long long)blockIdx.x * 256 + threadIdx.x;
  if (i >= total) return;
  long long r = i / cols;
  dst[i] = (r < rows) ? f2bf(src[i]) : (unsigned short)0;
}

__global__ void pad_f32(const float* __restrict__ src, float* __restrict__ dst, int n, int npad) {
  int i = blockIdx.x * 256 + threadIdx.x;
  if (i < npad) dst[i] = (i < n) ? src[i] : 0.0f;
}

// ---------------------------------------------------------------------------
// RoI align (FPN level select, sampling_ratio 2, 7x7 out) -> bf16 activations
// ---------------------------------------------------------------------------
__global__ __launch_bounds__(256)
void roi_align_kernel(const float* __restrict__ f0, const float* __restrict__ f1,
                      const float* __restrict__ f2, const float* __restrict__ f3,
                      const float* __restrict__ props, unsigned short* __restrict__ A) {
  int n = blockIdx.x;
  int tid = threadIdx.x;
  if (n >= NPROP) {                       // zero pad rows so GEMM pad output is benign
    for (int i = tid; i < D_IN; i += 256) A[(size_t)n * D_IN + i] = 0;
    return;
  }
  float x1 = props[n * 4 + 0], y1 = props[n * 4 + 1];
  float x2 = props[n * 4 + 2], y2 = props[n * 4 + 3];
  float wp = x2 - x1, hp = y2 - y1;
  float sc = sqrtf(fmaxf(wp * hp, 1e-6f));
  int lvl = (int)floorf(log2f(sc / 56.0f + 1e-6f));
  lvl = lvl < 0 ? 0 : (lvl > 3 ? 3 : lvl);
  const float* fp = (lvl == 0) ? f0 : (lvl == 1) ? f1 : (lvl == 2) ? f2 : f3;
  int H = 200 >> lvl, W = 336 >> lvl;
  float ss = 1.0f / (float)(4 << lvl);

  __shared__ float lx[PGRID], ly[PGRID];
  __shared__ int x0s[PGRID], x1s[PGRID], y0s[PGRID], y1s[PGRID];
  __shared__ int vxs[PGRID], vys[PGRID];
  if (tid < PGRID) {
    int p = tid;
    float pt = ((float)p + 0.5f) / (float)PGRID;
    float xs = x1 * ss - 0.5f + pt * (wp * ss);
    float ysv = y1 * ss - 0.5f + pt * (hp * ss);
    float xf = floorf(xs), yf = floorf(ysv);
    lx[p] = xs - xf; ly[p] = ysv - yf;
    int x0 = (int)xf; x0 = x0 < 0 ? 0 : (x0 > W - 1 ? W - 1 : x0);
    int y0 = (int)yf; y0 = y0 < 0 ? 0 : (y0 > H - 1 ? H - 1 : y0);
    x0s[p] = x0; x1s[p] = (x0 + 1 > W - 1) ? W - 1 : x0 + 1;
    y0s[p] = y0; y1s[p] = (y0 + 1 > H - 1) ? H - 1 : y0 + 1;
    vxs[p] = (xs >= -1.0f && xs <= (float)W);
    vys[p] = (ysv >= -1.0f && ysv <= (float)H);
  }
  __syncthreads();
  for (int i = tid; i < D_IN; i += 256) {
    int c = i / 49; int r = i - c * 49; int ph = r / 7; int pw = r - ph * 7;
    const float* base = fp + (size_t)c * H * W;
    float acc = 0.f;
#pragma unroll
    for (int sy2 = 0; sy2 < 2; ++sy2) {
      int py = ph * 2 + sy2;
#pragma unroll
      for (int sx2 = 0; sx2 < 2; ++sx2) {
        int px = pw * 2 + sx2;
        float v = 0.f;
        if (vys[py] && vxs[px]) {
          float a = ly[py], b = lx[px];
          const float* r0 = base + (size_t)y0s[py] * W;
          const float* r1 = base + (size_t)y1s[py] * W;
          float f00 = r0[x0s[px]], f01 = r0[x1s[px]];
          float f10 = r1[x0s[px]], f11 = r1[x1s[px]];
          v = (1.f - a) * ((1.f - b) * f00 + b * f01) + a * ((1.f - b) * f10 + b * f11);
        }
        acc += v;
      }
    }
    A[(size_t)n * D_IN + i] = f2bf(acc * 0.25f);
  }
}

// ---------------------------------------------------------------------------
// WMMA GEMM: out[M=1024, Nld] = A[M,K](bf16) * W[Nld,K]^T(bf16) + bias
// 256 threads = 8 waves; block tile 64x128; wave tile 16x64 (4 wmma accums).
// Double-buffered LDS filled by GLOBAL_LOAD_ASYNC_TO_LDS_B128 (ASYNCcnt),
// so iteration kt computes on buffer kt&1 while buffer (kt+1)&1 streams in.
// ---------------------------------------------------------------------------
__global__ __launch_bounds__(256)
void gemm_wmma(const unsigned short* __restrict__ A,
               const unsigned short* __restrict__ Wt,
               const float* __restrict__ bias,
               float* __restrict__ outF,
               unsigned short* __restrict__ outB,
               int K, int Nld, int doRelu) {
  __shared__ __attribute__((aligned(16))) unsigned short sA[2][64][40];   // 80B rows (bank-pad)
  __shared__ __attribute__((aligned(16))) unsigned short sB[2][128][40];
  int tid = threadIdx.x;
  int lane = tid & 31, wave = tid >> 5;
  int wr = wave & 3, wc = wave >> 2;          // wave's row tile / col half
  int l15 = lane & 15;
  int kb = (lane >> 4) * 8;                   // K-half select per ISA A/B layout
  int mb = blockIdx.x * 64;
  int nb = blockIdx.y * 128;

  union Frag { v16bf v; uint4 q[2]; };
  v8f acc[4] = {};

  int arow = tid >> 2, akc = (tid & 3) * 8;   // A stage: 64 rows x 32 k (1x16B per thread)
  int brow = tid >> 1, bkc = (tid & 1) * 16;  // B stage: 128 rows x 32 k (2x16B per thread)
  const int nsteps = K >> 5;

  // prologue: stream K-step 0 into buffer 0
  {
    async_copy16((unsigned)(size_t)&sA[0][arow][akc], A + (size_t)(mb + arow) * K + akc);
    async_copy16((unsigned)(size_t)&sB[0][brow][bkc], Wt + (size_t)(nb + brow) * K + bkc);
    async_copy16((unsigned)(size_t)&sB[0][brow][bkc + 8], Wt + (size_t)(nb + brow) * K + bkc + 8);
  }
  for (int kt = 0; kt < nsteps; ++kt) {
    int cur = kt & 1;
    if (kt + 1 < nsteps) {
      int nxt = cur ^ 1;
      int k1 = (kt + 1) << 5;
      async_copy16((unsigned)(size_t)&sA[nxt][arow][akc], A + (size_t)(mb + arow) * K + k1 + akc);
      async_copy16((unsigned)(size_t)&sB[nxt][brow][bkc], Wt + (size_t)(nb + brow) * K + k1 + bkc);
      async_copy16((unsigned)(size_t)&sB[nxt][brow][bkc + 8],
                   Wt + (size_t)(nb + brow) * K + k1 + bkc + 8);
      wait_async_le3();   // 3 newest may stay in flight; current buffer's copies are done
    } else {
      wait_async_0();
    }
    __syncthreads();      // all threads' copies for `cur` have landed in LDS
    Frag af;
    af.q[0] = *(const uint4*)&sA[cur][wr * 16 + l15][kb];
    af.q[1] = *(const uint4*)&sA[cur][wr * 16 + l15][kb + 16];
#pragma unroll
    for (int t = 0; t < 4; ++t) {
      Frag bf;
      int nrow = wc * 64 + t * 16 + l15;
      bf.q[0] = *(const uint4*)&sB[cur][nrow][kb];
      bf.q[1] = *(const uint4*)&sB[cur][nrow][kb + 16];
      acc[t] = __builtin_amdgcn_wmma_f32_16x16x32_bf16(false, af.v, false, bf.v,
                                                       (short)0, acc[t], false, false);
    }
    __syncthreads();      // `cur` buffer may be overwritten next iteration
  }
  // C/D layout: element v of lane l -> M = v + 8*(l>=16), N = l&15
#pragma unroll
  for (int t = 0; t < 4; ++t) {
    int col = nb + wc * 64 + t * 16 + l15;
    float bv = bias[col];
#pragma unroll
    for (int v = 0; v < 8; ++v) {
      int row = mb + wr * 16 + v + kb;   // kb == 8*(lane>=16)
      float x = acc[t][v] + bv;
      if (doRelu) x = fmaxf(x, 0.f);
      if (outF) outF[(size_t)row * Nld + col] = x;
      if (outB) outB[(size_t)row * Nld + col] = f2bf(x);
    }
  }
}

// ---------------------------------------------------------------------------
// Softmax + delta2bbox, transposed to class-major for NMS
// ---------------------------------------------------------------------------
__global__ __launch_bounds__(256)
void postprocess(const float* __restrict__ cls, const float* __restrict__ reg,
                 const float* __restrict__ props,
                 float* __restrict__ scoresT, float* __restrict__ boxesT) {
  int n = blockIdx.x * 256 + threadIdx.x;
  if (n >= NPROP) return;
  const float* cs = cls + (size_t)n * CLSP;
  float mx = cs[0];
  for (int j = 1; j < NCLS; ++j) mx = fmaxf(mx, cs[j]);
  float sum = 0.f;
  for (int j = 0; j < NCLS; ++j) sum += expf(cs[j] - mx);
  float inv = 1.0f / sum;
  float x1 = props[n * 4 + 0], y1 = props[n * 4 + 1];
  float x2 = props[n * 4 + 2], y2 = props[n * 4 + 3];
  float px = (x1 + x2) * 0.5f, py = (y1 + y2) * 0.5f;
  float pw = x2 - x1, ph = y2 - y1;
  const float MR = 4.135166556742356f;       // |log(16/1000)|
  const float* rg = reg + (size_t)n * REGP;
  for (int c = 0; c < 80; ++c) {
    float dx = rg[c * 4 + 0] * 0.1f, dy = rg[c * 4 + 1] * 0.1f;
    float dw = fminf(fmaxf(rg[c * 4 + 2] * 0.2f, -MR), MR);
    float dh = fminf(fmaxf(rg[c * 4 + 3] * 0.2f, -MR), MR);
    float gw = pw * expf(dw), gh = ph * expf(dh);
    float gx = px + pw * dx, gy = py + ph * dy;
    size_t o = (size_t)c * NPROP + n;
    scoresT[o] = expf(cs[c] - mx) * inv;
    boxesT[o * 4 + 0] = gx - gw * 0.5f;
    boxesT[o * 4 + 1] = gy - gh * 0.5f;
    boxesT[o * 4 + 2] = gx + gw * 0.5f;
    boxesT[o * 4 + 3] = gy + gh * 0.5f;
  }
}

// ---------------------------------------------------------------------------
// Per-class NMS: bitonic sort (desc) of 1024, then sequential greedy suppress
// ---------------------------------------------------------------------------
__global__ __launch_bounds__(256)
void nms_class(const float* __restrict__ scoresT, const float* __restrict__ boxesT,
               float* __restrict__ sortedB, float* __restrict__ sflat) {
  int c = blockIdx.x;
  int tid = threadIdx.x;
  __shared__ float ss[1024];
  __shared__ int   si[1024];
  __shared__ float bx[1024][4];
  __shared__ float ar[1024];
  __shared__ unsigned char kp[1024];
  for (int i = tid; i < 1024; i += 256) {
    ss[i] = (i < NPROP) ? scoresT[(size_t)c * NPROP + i] : -1e30f;
    si[i] = i;
  }
  __syncthreads();
  for (int k = 2; k <= 1024; k <<= 1) {
    for (int j = k >> 1; j > 0; j >>= 1) {
      for (int i = tid; i < 1024; i += 256) {
        int ixj = i ^ j;
        if (ixj > i) {
          bool desc = ((i & k) == 0);
          float a = ss[i], b = ss[ixj];
          bool sw = desc ? (a < b) : (a > b);
          if (sw) { ss[i] = b; ss[ixj] = a; int t = si[i]; si[i] = si[ixj]; si[ixj] = t; }
        }
      }
      __syncthreads();
    }
  }
  for (int i = tid; i < 1024; i += 256) {
    int idx = si[i];
    float b0 = 0, b1 = 0, b2 = 0, b3 = 0;
    if (idx < NPROP) {
      size_t o = ((size_t)c * NPROP + idx) * 4;
      b0 = boxesT[o]; b1 = boxesT[o + 1]; b2 = boxesT[o + 2]; b3 = boxesT[o + 3];
    }
    bx[i][0] = b0; bx[i][1] = b1; bx[i][2] = b2; bx[i][3] = b3;
    ar[i] = fmaxf(b2 - b0, 0.f) * fmaxf(b3 - b1, 0.f);
    kp[i] = (unsigned char)((i < NPROP) && (ss[i] > 0.05f));
  }
  __syncthreads();
  for (int i = 0; i < NPROP; ++i) {
    if (kp[i]) {
      float ax1 = bx[i][0], ay1 = bx[i][1], ax2 = bx[i][2], ay2 = bx[i][3], aa = ar[i];
      for (int j = i + 1 + tid; j < NPROP; j += 256) {
        if (kp[j]) {
          float iw = fmaxf(fminf(ax2, bx[j][2]) - fmaxf(ax1, bx[j][0]), 0.f);
          float ih = fmaxf(fminf(ay2, bx[j][3]) - fmaxf(ay1, bx[j][1]), 0.f);
          float inter = iw * ih;
          float iou = inter / fmaxf(aa + ar[j] - inter, 1e-8f);
          if (iou > 0.5f) kp[j] = 0;
        }
      }
    }
    __syncthreads();
  }
  for (int i = tid; i < NPROP; i += 256) {
    sflat[(size_t)c * NPROP + i] = kp[i] ? ss[i] : -1.0f;
    size_t o = ((size_t)c * NPROP + i) * 4;
    sortedB[o] = bx[i][0]; sortedB[o + 1] = bx[i][1];
    sortedB[o + 2] = bx[i][2]; sortedB[o + 3] = bx[i][3];
  }
}

// ---------------------------------------------------------------------------
// Top-100 over masked scores; emit [num_det | boxes*400 | scores*100 | classes*100]
// ---------------------------------------------------------------------------
__global__ __launch_bounds__(256)
void topk_kernel(float* __restrict__ sflat, const float* __restrict__ sortedB,
                 float* __restrict__ out) {
  int tid = threadIdx.x;
  __shared__ float rv[256]; __shared__ int ri[256];
  __shared__ float bvs[100]; __shared__ int bis[100];
  const int TOT = 80 * NPROP;
  for (int it = 0; it < 100; ++it) {
    float v = -1e30f; int id = 0;
    for (int i = tid; i < TOT; i += 256) {
      float s = sflat[i];
      if (s > v) { v = s; id = i; }
    }
    rv[tid] = v; ri[tid] = id;
    __syncthreads();
    for (int s = 128; s > 0; s >>= 1) {
      if (tid < s) {
        float ov = rv[tid + s]; int oi = ri[tid + s];
        if (ov > rv[tid] || (ov == rv[tid] && oi < ri[tid])) { rv[tid] = ov; ri[tid] = oi; }
      }
      __syncthreads();
    }
    if (tid == 0) {
      bvs[it] = rv[0]; bis[it] = ri[0];
      sflat[ri[0]] = -2e30f;
      __threadfence();
    }
    __syncthreads();
  }
  for (int it = tid; it < 100; it += 256) {
    float v = bvs[it]; int idx = bis[it];
    bool valid = v > 0.0f;
    out[401 + it] = valid ? v : 0.0f;
    out[501 + it] = valid ? (float)(idx / NPROP) : -1.0f;
    size_t o = (size_t)idx * 4;
    for (int k = 0; k < 4; ++k) out[1 + it * 4 + k] = valid ? sortedB[o + k] : 0.0f;
  }
  if (tid == 0) {
    int nd = 0;
    for (int i = 0; i < 100; ++i) nd += (bvs[i] > 0.0f);
    out[0] = (float)nd;
  }
}

// ---------------------------------------------------------------------------
extern "C" void kernel_launch(void* const* d_in, const int* in_sizes, int n_in,
                              void* d_out, int out_size, void* d_ws, size_t ws_size,
                              hipStream_t stream) {
  const float* f0   = (const float*)d_in[0];
  const float* f1   = (const float*)d_in[1];
  const float* f2   = (const float*)d_in[2];
  const float* f3   = (const float*)d_in[3];
  const float* prop = (const float*)d_in[4];
  const float* w1   = (const float*)d_in[5];
  const float* b1   = (const float*)d_in[6];
  const float* w2   = (const float*)d_in[7];
  const float* b2   = (const float*)d_in[8];
  const float* wc   = (const float*)d_in[9];
  const float* bc   = (const float*)d_in[10];
  const float* wr   = (const float*)d_in[11];
  const float* br   = (const float*)d_in[12];

  char* ws = (char*)d_ws;
  size_t off = 0;
  auto alloc = [&](size_t bytes) -> char* {
    char* p = ws + off;
    off += (bytes + 255) & ~(size_t)255;
    return p;
  };
  unsigned short* A0  = (unsigned short*)alloc((size_t)MPAD * D_IN * 2);
  unsigned short* W1b = (unsigned short*)alloc((size_t)FC * D_IN * 2);
  unsigned short* H1  = (unsigned short*)alloc((size_t)MPAD * FC * 2);
  unsigned short* W2b = (unsigned short*)alloc((size_t)FC * FC * 2);
  unsigned short* H2  = (unsigned short*)alloc((size_t)MPAD * FC * 2);
  unsigned short* WCb = (unsigned short*)alloc((size_t)CLSP * FC * 2);
  unsigned short* WRb = (unsigned short*)alloc((size_t)REGP * FC * 2);
  float* CLS  = (float*)alloc((size_t)MPAD * CLSP * 4);
  float* REG  = (float*)alloc((size_t)MPAD * REGP * 4);
  float* B1p  = (float*)alloc((size_t)FC * 4);
  float* B2p  = (float*)alloc((size_t)FC * 4);
  float* BCp  = (float*)alloc((size_t)CLSP * 4);
  float* BRp  = (float*)alloc((size_t)REGP * 4);
  float* SCT  = (float*)alloc((size_t)80 * NPROP * 4);
  float* BXT  = (float*)alloc((size_t)80 * NPROP * 16);
  float* SRT  = (float*)alloc((size_t)80 * NPROP * 16);
  float* SFL  = (float*)alloc((size_t)80 * NPROP * 4);
  (void)ws_size; (void)in_sizes; (void)n_in; (void)out_size;

  // 1) weight / bias conversion + padding
  {
    long long t1 = (long long)FC * D_IN;
    cvt_pad_bf16<<<(unsigned)((t1 + 255) / 256), 256, 0, stream>>>(w1, W1b, FC, D_IN, t1);
    long long t2 = (long long)FC * FC;
    cvt_pad_bf16<<<(unsigned)((t2 + 255) / 256), 256, 0, stream>>>(w2, W2b, FC, FC, t2);
    long long t3 = (long long)CLSP * FC;
    cvt_pad_bf16<<<(unsigned)((t3 + 255) / 256), 256, 0, stream>>>(wc, WCb, NCLS, FC, t3);
    long long t4 = (long long)REGP * FC;
    cvt_pad_bf16<<<(unsigned)((t4 + 255) / 256), 256, 0, stream>>>(wr, WRb, 320, FC, t4);
    pad_f32<<<(FC + 255) / 256, 256, 0, stream>>>(b1, B1p, FC, FC);
    pad_f32<<<(FC + 255) / 256, 256, 0, stream>>>(b2, B2p, FC, FC);
    pad_f32<<<1, 256, 0, stream>>>(bc, BCp, NCLS, CLSP);
    pad_f32<<<2, 256, 0, stream>>>(br, BRp, 320, REGP);
  }
  // 2) RoI align into bf16 activations (rows >= 1000 zeroed)
  roi_align_kernel<<<MPAD, 256, 0, stream>>>(f0, f1, f2, f3, prop, A0);
  // 3) FC1: [1024,12544] x [12544,1024] -> bf16 + relu
  gemm_wmma<<<dim3(MPAD / 64, FC / 128), 256, 0, stream>>>(A0, W1b, B1p, nullptr, H1, D_IN, FC, 1);
  // 4) FC2
  gemm_wmma<<<dim3(MPAD / 64, FC / 128), 256, 0, stream>>>(H1, W2b, B2p, nullptr, H2, FC, FC, 1);
  // 5) cls / reg heads -> f32
  gemm_wmma<<<dim3(MPAD / 64, CLSP / 128), 256, 0, stream>>>(H2, WCb, BCp, CLS, nullptr, FC, CLSP, 0);
  gemm_wmma<<<dim3(MPAD / 64, REGP / 128), 256, 0, stream>>>(H2, WRb, BRp, REG, nullptr, FC, REGP, 0);
  // 6) softmax + delta2bbox (class-major)
  postprocess<<<(NPROP + 255) / 256, 256, 0, stream>>>(CLS, REG, prop, SCT, BXT);
  // 7) per-class NMS
  nms_class<<<80, 256, 0, stream>>>(SCT, BXT, SRT, SFL);
  // 8) top-100 + output assembly
  topk_kernel<<<1, 256, 0, stream>>>(SFL, SRT, (float*)d_out);
}
